// MRU_6863357739205
// MI455X (gfx1250) — compile-verified, matrix operand order checked by reading, not run on hardware
//
#include <hip/hip_runtime.h>
#include <cmath>

typedef _Float16 f16;
typedef __attribute__((ext_vector_type(16))) _Float16 v16h;
typedef __attribute__((ext_vector_type(8)))  float    v8f;

#define NB      64
#define LSEQ    1000
#define DIM     300
#define KT      10          // K tiles of 32 (K padded 300 -> 320)
#define NT      19          // N tiles of 16 (N padded 300 -> 304)
#define FRAGE   512         // f16 elements per 16x32 (or 32x16) fragment
#define TOTMT   7560        // total M-tiles over all 5 ranges
#define TOTROWS 120960      // total rows over all 5 ranges

// ---------------------------------------------------------------------------
// fragment element mapping (CDNA5 ISA §7.12.2, 16-bit A 16x32):
// lane 0-15  : M=lane,  VGPR0..3 -> K 0..7 (2/vgpr),  VGPR4..7 -> K 16..23
// lane 16-31 : M=lane-16, VGPR0..3 -> K 8..15,        VGPR4..7 -> K 24..31
// B (32x16) mirrors with N in place of M.
// element p (0..15) of a lane's v16h: vgpr v=p/2, half slot=p&1
__device__ __forceinline__ int frag_klocal(int lane, int p) {
    const int v = p >> 1, slot = p & 1, halfsel = lane >> 4;
    return ((v & 4) ? 16 : 0) + halfsel * 8 + (v & 3) * 2 + slot;
}

// ---------------------------------------------------------------------------
// Kernel 1: convert f32 weights -> pre-swizzled f16 WMMA-B fragments.
// grid.x = 7 * NT * KT, 256 threads
__global__ void conv_weights_kernel(const float* __restrict__ wce,
                                    const float* __restrict__ wz,
                                    const float* __restrict__ wo,
                                    f16* __restrict__ wh) {
    const int bid = blockIdx.x;
    const int mat = bid / (NT * KT);
    const int rem = bid % (NT * KT);
    const int nt  = rem / KT;
    const int kt  = rem % KT;
    const float* W = (mat < 5) ? (wce + (size_t)mat * DIM * DIM)
                               : ((mat == 5) ? wz : wo);
    for (int e = threadIdx.x; e < FRAGE; e += 256) {
        const int lane = e >> 4, p = e & 15;
        const int nl = lane & 15;
        const int kl = frag_klocal(lane, p);
        const int k = kt * 32 + kl, n = nt * 16 + nl;
        const float val = (k < DIM && n < DIM) ? W[k * DIM + n] : 0.f;
        wh[(size_t)bid * FRAGE + e] = (f16)val;
    }
}

// ---------------------------------------------------------------------------
// Kernel 2: embedding gather + segment sums + A-fragmentize.
// One block per 16-row M-tile of each range's flattened [B*L/r] row space.
// grid.x = TOTMT, 256 threads; LDS: 16 x 320 f32 accumulator.
__global__ void gather_frag_kernel(const int* __restrict__ article,
                                   const float* __restrict__ emb,
                                   f16* __restrict__ frag) {
    __shared__ float acc[16 * 320];
    __shared__ int   toks[16];
    const int bid = blockIdx.x;
    const int tid = threadIdx.x;

    int ri, mlocal;
    if      (bid < 4000) { ri = 0; mlocal = bid;        }
    else if (bid < 6000) { ri = 1; mlocal = bid - 4000; }
    else if (bid < 7000) { ri = 2; mlocal = bid - 6000; }
    else if (bid < 7400) { ri = 3; mlocal = bid - 7000; }
    else                 { ri = 4; mlocal = bid - 7400; }
    const int r  = (ri == 0) ? 1 : (ri == 1) ? 2 : (ri == 2) ? 4 : (ri == 3) ? 10 : 25;
    const int Lr = LSEQ / r;

    for (int i = tid; i < 16 * 320; i += 256) acc[i] = 0.f;
    __syncthreads();

    const int segbase = mlocal * 16;
    for (int j = 0; j < r; ++j) {
        if (tid < 16) {
            const int q  = segbase + tid;
            const int b  = q / Lr;
            const int ls = q - b * Lr;
            toks[tid] = article[b * LSEQ + ls * r + j];
        }
        __syncthreads();
        for (int i = tid; i < 16 * DIM; i += 256) {
            const int s = i / DIM;
            const int d = i - s * DIM;
            acc[s * 320 + d] += emb[(size_t)toks[s] * DIM + d];
        }
        __syncthreads();
    }

    // emit pre-swizzled f16 A fragments (zeros in K-pad come from LDS init)
    const int wave = tid >> 5;
    const int lane = tid & 31;
    const int m    = lane & 15;
    for (int kt = wave; kt < KT; kt += 8) {
        v16h av;
#pragma unroll
        for (int p = 0; p < 16; ++p) {
            const int kl = frag_klocal(lane, p);
            av[p] = (f16)acc[m * 320 + kt * 32 + kl];
        }
        *(v16h*)(frag + ((size_t)bid * KT + kt) * FRAGE + lane * 16) = av;
    }
}

// ---------------------------------------------------------------------------
// Kernel 3: WMMA GEMM, Y = act(X @ W) (+bias placement per mode).
// mode 0: relu(x)+bias (contract-expand)   mode 1: tanh(x)+bias (z / o)
// One wave per 16-row M-tile; 8 waves / block.
__global__ void __launch_bounds__(256)
gemm_wmma_kernel(const f16* __restrict__ afrag,
                 const f16* __restrict__ wh,
                 const float* __restrict__ bias,
                 f16* __restrict__ y,
                 int nmtiles, int mode) {
    const int wave  = threadIdx.x >> 5;
    const int lane  = threadIdx.x & 31;
    const int mtile = blockIdx.x * 8 + wave;
    if (mtile >= nmtiles) return;

    v16h a[KT];
#pragma unroll
    for (int kt = 0; kt < KT; ++kt)
        a[kt] = *(const v16h*)(afrag + ((size_t)mtile * KT + kt) * FRAGE + lane * 16);

    const int nlane = lane & 15;
    const int rbase = mtile * 16 + ((lane < 16) ? 0 : 8);

    for (int nt = 0; nt < NT; ++nt) {
        v8f acc = {};
#pragma unroll
        for (int kt = 0; kt < KT; ++kt) {
            const v16h bf = *(const v16h*)(wh + ((size_t)nt * KT + kt) * FRAGE + lane * 16);
            acc = __builtin_amdgcn_wmma_f32_16x16x32_f16(
                false, a[kt], false, bf, (short)0, acc, false, false);
        }
        const int ncol = nt * 16 + nlane;
        if (ncol < DIM) {
            const float bv = bias[ncol];
#pragma unroll
            for (int v = 0; v < 8; ++v) {
                float x = acc[v];
                x = mode ? (tanhf(x) + bv) : (fmaxf(x, 0.f) + bv);
                y[(size_t)(rbase + v) * DIM + ncol] = (f16)x;
            }
        }
    }
}

// ---------------------------------------------------------------------------
// Kernel 4: pointwise multi-range gate MLP (5 -> 3 -> 1 across ranges).
__global__ void gate_kernel(const f16* __restrict__ tbuf,
                            const float* __restrict__ w1,
                            const float* __restrict__ b1,
                            const float* __restrict__ w2,
                            const float* __restrict__ b2,
                            f16* __restrict__ gate) {
    const size_t e = (size_t)blockIdx.x * blockDim.x + threadIdx.x;
    if (e >= (size_t)NB * LSEQ * DIM) return;
    const int    d  = (int)(e % DIM);
    const size_t bl = e / DIM;
    const int    l  = (int)(bl % LSEQ);
    const int    b  = (int)(bl / LSEQ);

    const int rr[5]     = {1, 2, 4, 10, 25};
    const int rowoff[5] = {0, 64000, 96000, 112000, 118400};
    float pre[5];
#pragma unroll
    for (int s = 0; s < 5; ++s) {
        const int Lr = LSEQ / rr[s];
        const size_t row = (size_t)rowoff[s] + (size_t)b * Lr + l / rr[s];
        pre[s] = (float)tbuf[row * DIM + d] * (1.f / (float)rr[s]);
    }
    float g = b2[d];
#pragma unroll
    for (int t = 0; t < 3; ++t) {
        float h = b1[t * DIM + d];
#pragma unroll
        for (int s = 0; s < 5; ++s) h += w1[t * 5 + s] * pre[s];
        g += w2[t] * fmaxf(h, 0.f);
    }
    gate[e] = (f16)fmaxf(g, 0.f);
}

// ---------------------------------------------------------------------------
// Kernel 5: sequential gated recurrence over L, one thread per (b, d).
__global__ void scan_kernel(const f16* __restrict__ gate,
                            const f16* __restrict__ zb,
                            const f16* __restrict__ ob,
                            float* __restrict__ out) {
    const int t = blockIdx.x * blockDim.x + threadIdx.x;  // 0 .. B*DIM-1
    const int b = t / DIM;
    const int d = t - b * DIM;
    size_t idx = (size_t)b * LSEQ * DIM + d;
    float c = 0.f;
    for (int l = 0; l < LSEQ; ++l, idx += DIM) {
        if (l + 16 < LSEQ) {
            __builtin_prefetch(gate + idx + 16 * DIM, 0, 0);
            __builtin_prefetch(zb   + idx + 16 * DIM, 0, 0);
            __builtin_prefetch(ob   + idx + 16 * DIM, 0, 0);
        }
        const float g = (float)gate[idx];
        const float z = (float)zb[idx];
        const float o = (float)ob[idx];
        c = g * c + (1.f - g) * z;
        out[idx] = o * c;
    }
}

// ---------------------------------------------------------------------------
extern "C" void kernel_launch(void* const* d_in, const int* in_sizes, int n_in,
                              void* d_out, int out_size, void* d_ws, size_t ws_size,
                              hipStream_t stream) {
    const int*   article = (const int*)d_in[0];
    const float* emb = (const float*)d_in[1];
    const float* wce = (const float*)d_in[2];
    const float* bce = (const float*)d_in[3];
    const float* w1  = (const float*)d_in[4];
    const float* b1  = (const float*)d_in[5];
    const float* w2  = (const float*)d_in[6];
    const float* b2  = (const float*)d_in[7];
    const float* wz  = (const float*)d_in[8];
    const float* bz  = (const float*)d_in[9];
    const float* wo  = (const float*)d_in[10];
    const float* bo  = (const float*)d_in[11];
    float* out = (float*)d_out;

    // workspace layout (bytes), all 256B-aligned
    constexpr size_t WH_BYTES   = 7ull * NT * KT * FRAGE * sizeof(f16);      //  1.36 MB
    constexpr size_t FRAG_BYTES = (size_t)TOTMT * KT * FRAGE * sizeof(f16);  // 77.4  MB
    constexpr size_t T_BYTES    = (size_t)TOTROWS * DIM * sizeof(f16);       // 72.6  MB
    constexpr size_t ZO_BYTES   = (size_t)NB * LSEQ * DIM * sizeof(f16);     // 38.4  MB
    char* ws = (char*)d_ws;
    f16* wh   = (f16*)(ws);
    f16* frag = (f16*)(ws + WH_BYTES);
    f16* tbuf = (f16*)(ws + WH_BYTES + FRAG_BYTES);
    f16* zbuf = (f16*)(ws + WH_BYTES + FRAG_BYTES + T_BYTES);
    f16* obuf = (f16*)(ws + WH_BYTES + FRAG_BYTES + T_BYTES + ZO_BYTES);
    f16* gatebuf = frag;  // fragments dead by the time gate is produced

    // 1) weight swizzle (7 matrices)
    conv_weights_kernel<<<7 * NT * KT, 256, 0, stream>>>(wce, wz, wo, wh);

    // 2) gather + segment sums + A-fragments for all 5 ranges
    gather_frag_kernel<<<TOTMT, 256, 0, stream>>>(article, emb, frag);

    // 3) seven WMMA GEMMs
    const int mtiles[5] = {4000, 2000, 1000, 400, 160};
    const int mtoff[5]  = {0, 4000, 6000, 7000, 7400};
    const int rowoff[5] = {0, 64000, 96000, 112000, 118400};
    for (int i = 0; i < 5; ++i) {
        gemm_wmma_kernel<<<mtiles[i] / 8, 256, 0, stream>>>(
            frag + (size_t)mtoff[i] * KT * FRAGE,
            wh + (size_t)i * NT * KT * FRAGE,
            bce + (size_t)i * DIM,
            tbuf + (size_t)rowoff[i] * DIM,
            mtiles[i], /*mode=*/0);
    }
    gemm_wmma_kernel<<<4000 / 8, 256, 0, stream>>>(
        frag, wh + 5ull * NT * KT * FRAGE, bz, zbuf, 4000, /*mode=*/1);
    gemm_wmma_kernel<<<4000 / 8, 256, 0, stream>>>(
        frag, wh + 6ull * NT * KT * FRAGE, bo, obuf, 4000, /*mode=*/1);

    // 4) gate MLP (19.2M elements)
    gate_kernel<<<(NB * LSEQ * DIM) / 256, 256, 0, stream>>>(
        tbuf, w1, b1, w2, b2, gatebuf);

    // 5) gated recurrence
    scan_kernel<<<(NB * DIM) / 256, 256, 0, stream>>>(gatebuf, zbuf, obuf, out);
}
